// Attention_65996467470822
// MI455X (gfx1250) — compile-verified, matrix operand order checked by reading, not run on hardware
//
#include <hip/hip_runtime.h>
#include <hip/hip_bf16.h>
#include <math.h>

// ---------------------------------------------------------------------------
// MI455X (gfx1250) attention block:
//   qkv = X @ Wqkv ; rotary(q,k) ; causal flash attention ; out = A @ Wout + b
// All matmuls use v_wmma_f32_16x16x32_bf16 (f32 accumulate), wave32.
// ---------------------------------------------------------------------------

typedef __bf16 v16bf __attribute__((ext_vector_type(16)));
typedef float  v8f   __attribute__((ext_vector_type(8)));

#define B_    2
#define N_    2048
#define QD    1024
#define H_    16
#define DH    64
#define INNER 1024
#define NQKV  3072
#define ROWS  (B_ * N_)   // 4096

union Frag {
  v16bf f;
  uint4 u[2];
};

#define WMMA_BF16(A, Bm, C) \
  __builtin_amdgcn_wmma_f32_16x16x32_bf16(false, (A), false, (Bm), (short)0, (C), false, false)

// ---------------------------------------------------------------------------
// fp32 -> bf16 convert (plain) and convert+transpose (for weights)
// ---------------------------------------------------------------------------
__global__ void cvt_kernel(const float* __restrict__ s, __bf16* __restrict__ d, int n) {
  int i = blockIdx.x * blockDim.x + threadIdx.x;
  if (i < n) d[i] = (__bf16)s[i];
}

// s is [K][Nn] row-major fp32; d is [Nn][K] row-major bf16
__global__ void cvt_t_kernel(const float* __restrict__ s, __bf16* __restrict__ d,
                             int K, int Nn) {
  int i = blockIdx.x * blockDim.x + threadIdx.x;
  if (i < K * Nn) {
    int k = i / Nn;
    int n = i - k * Nn;
    d[(size_t)n * K + k] = (__bf16)s[i];
  }
}

// ---------------------------------------------------------------------------
// Tiled GEMM:  C[M][N] = A[M][K] * Bt[N][K]^T   (bf16 in, f32 accumulate)
// Block: 256 threads (8 waves), 128x128 C tile, K-slab 64 (16 WMMAs/barrier).
// Wave grid 2x4; each wave owns a 64x32 sub-tile = 4x2 WMMA accumulators.
// OUTF32==1: add bias, store f32.  OUTF32==0: store bf16.
// ---------------------------------------------------------------------------
__device__ __forceinline__ Frag ld_frag_lds64(const __bf16* base, int rowBase,
                                              int kcol, int lane) {
  // base is a [128][64] bf16 LDS tile; returns a 16x32 frag at (rowBase, kcol).
  Frag r;
  const int kq = (lane < 16) ? 0 : 8;
  const __bf16* p = base + (size_t)(rowBase + (lane & 15)) * 64 + kcol;
  r.u[0] = *(const uint4*)(p + kq);
  r.u[1] = *(const uint4*)(p + kq + 16);
  return r;
}

template <int OUTF32>
__global__ __launch_bounds__(256)
void gemm_bf16_kernel(const __bf16* __restrict__ A,   // [M][K]
                      const __bf16* __restrict__ Bt,  // [N][K]
                      int M, int N, int K,
                      const float* __restrict__ bias, // nullable
                      float* __restrict__ Cf,
                      __bf16* __restrict__ Cb) {
  __shared__ __bf16 As[128 * 64];   // 16 KB
  __shared__ __bf16 Bs[128 * 64];   // 16 KB

  const int tid  = threadIdx.x;
  const int lane = tid & 31;
  const int w    = tid >> 5;   // 0..7
  const int wm   = w >> 2;     // 0..1 -> 64 rows each
  const int wn   = w & 3;      // 0..3 -> 32 cols each

  const int m0 = blockIdx.y * 128;
  const int n0 = blockIdx.x * 128;

  v8f acc[4][2];
#pragma unroll
  for (int i = 0; i < 4; i++)
#pragma unroll
    for (int j = 0; j < 2; j++)
      acc[i][j] = (v8f){0.f, 0.f, 0.f, 0.f, 0.f, 0.f, 0.f, 0.f};

  for (int k0 = 0; k0 < K; k0 += 64) {
    // Stage 128x64 slabs of A and Bt into LDS (both are [row][k] layout).
#pragma unroll
    for (int c = tid; c < 1024; c += 256) {
      const int row = c >> 3;
      const int cc  = (c & 7) * 8;
      *(uint4*)&As[row * 64 + cc] = *(const uint4*)(A  + (size_t)(m0 + row) * K + k0 + cc);
      *(uint4*)&Bs[row * 64 + cc] = *(const uint4*)(Bt + (size_t)(n0 + row) * K + k0 + cc);
    }
    // Prefetch next K slab into L2 while we compute (global_prefetch_b8).
    if (k0 + 64 < K) {
      __builtin_prefetch(A  + (size_t)(m0 + (tid >> 1)) * K + k0 + 64 + (tid & 1) * 32, 0, 1);
      __builtin_prefetch(Bt + (size_t)(n0 + (tid >> 1)) * K + k0 + 64 + (tid & 1) * 32, 0, 1);
    }
    __syncthreads();

#pragma unroll
    for (int ks = 0; ks < 2; ks++) {
      Frag af[4], bfr[2];
#pragma unroll
      for (int mt = 0; mt < 4; mt++)
        af[mt] = ld_frag_lds64(As, wm * 64 + mt * 16, ks * 32, lane);
#pragma unroll
      for (int nt = 0; nt < 2; nt++)
        bfr[nt] = ld_frag_lds64(Bs, wn * 32 + nt * 16, ks * 32, lane);

#pragma unroll
      for (int mt = 0; mt < 4; mt++)
#pragma unroll
        for (int nt = 0; nt < 2; nt++)
          acc[mt][nt] = WMMA_BF16(af[mt].f, bfr[nt].f, acc[mt][nt]);
    }
    __syncthreads();
  }

  // Epilogue. C layout: slot i -> M = i (+8 for upper lane half), N = lane&15.
  const int mro  = (lane >> 4) << 3;
  const int ncol = lane & 15;
#pragma unroll
  for (int mt = 0; mt < 4; mt++) {
#pragma unroll
    for (int nt = 0; nt < 2; nt++) {
      const int gn = n0 + wn * 32 + nt * 16 + ncol;
      const float badd = (OUTF32 && bias) ? bias[gn] : 0.f;
#pragma unroll
      for (int i = 0; i < 8; i++) {
        const int gm = m0 + wm * 64 + mt * 16 + mro + i;
        if (OUTF32) Cf[(size_t)gm * N + gn] = acc[mt][nt][i] + badd;
        else        Cb[(size_t)gm * N + gn] = (__bf16)acc[mt][nt][i];
      }
    }
  }
}

// ---------------------------------------------------------------------------
// Rotary + split: qkv[b,n,3*1024] -> Q[b,h,n,d] (rotated, PRE-SCALED by
// 1/sqrt(DH)), K[b,h,n,d] (rotated), V transposed -> Vt[b,h,d,n].
// One thread per (b,h,n,d-pair).
// ---------------------------------------------------------------------------
__global__ void rotary_split_kernel(const __bf16* __restrict__ qkv,
                                    __bf16* __restrict__ Q,
                                    __bf16* __restrict__ Kc,
                                    __bf16* __restrict__ Vt) {
  const int t = blockIdx.x * blockDim.x + threadIdx.x;
  if (t >= B_ * H_ * N_ * (DH / 2)) return;
  const int d2 = t & 31;            // pair index 0..31
  const int n  = (t >> 5) & (N_ - 1);
  const int h  = (t >> 16) & (H_ - 1);
  const int b  = t >> 20;

  const float inv = __powf(10000.f, -(float)(2 * d2) / 64.f);
  const float ang = (float)n * inv;
  const float cs = __cosf(ang), sn = __sinf(ang);
  const float qscale = 0.125f;      // 1/sqrt(64), folded into Q

  const size_t row = (size_t)(b * N_ + n) * NQKV;
  const int    qc  = h * DH + 2 * d2;

  const float q0 = (float)qkv[row + qc],             q1 = (float)qkv[row + qc + 1];
  const float k0 = (float)qkv[row + INNER + qc],     k1 = (float)qkv[row + INNER + qc + 1];
  const float v0 = (float)qkv[row + 2 * INNER + qc], v1 = (float)qkv[row + 2 * INNER + qc + 1];

  const size_t qk = ((size_t)(b * H_ + h) * N_ + n) * DH + 2 * d2;
  Q[qk]      = (__bf16)((q0 * cs - q1 * sn) * qscale);
  Q[qk + 1]  = (__bf16)((q1 * cs + q0 * sn) * qscale);
  Kc[qk]     = (__bf16)(k0 * cs - k1 * sn);
  Kc[qk + 1] = (__bf16)(k1 * cs + k0 * sn);

  const size_t vt = ((size_t)(b * H_ + h) * DH + 2 * d2) * N_ + n;
  Vt[vt]      = (__bf16)v0;
  Vt[vt + N_] = (__bf16)v1;
}

// ---------------------------------------------------------------------------
// Flash attention, one wave per 16-row q tile.  32-key steps.
//   S(16x32) via 4 WMMA (contraction d=64, scale pre-folded into Q),
//   online softmax with shfl_xor reductions inside each 16-lane half,
//   P transposed via 1KB LDS into an A-frag, O += P @ V via 4 WMMA
//   (contraction keys=32).  K/V re-reads across q-tiles hit the 192MB L2.
// Output written as [b, n, h*64+d] bf16 (ready for the out-projection GEMM).
// ---------------------------------------------------------------------------
__global__ __launch_bounds__(32)
void flash_attn_kernel(const __bf16* __restrict__ Q,
                       const __bf16* __restrict__ Kc,
                       const __bf16* __restrict__ Vt,
                       __bf16* __restrict__ O,
                       const int* __restrict__ is_causal) {
  __shared__ __bf16 Plds[16 * 32];

  const int lane = threadIdx.x & 31;
  const int qt = blockIdx.x;   // 0..127
  const int h  = blockIdx.y;
  const int b  = blockIdx.z;
  const bool causal = (*is_causal) != 0;
  const int q0 = qt * 16;

  const __bf16* Qp = Q  + (size_t)(b * H_ + h) * N_ * DH;
  const __bf16* Kp = Kc + (size_t)(b * H_ + h) * N_ * DH;
  const __bf16* Vp = Vt + (size_t)(b * H_ + h) * DH * N_;

  const int rl  = lane & 15;
  const int kq  = (lane < 16) ? 0 : 8;
  const int mro = (lane >> 4) << 3;

  // Q A-frags for d=0..31 and d=32..63 (held in registers for whole loop).
  Frag qa[2];
  {
    const __bf16* qr = Qp + (size_t)(q0 + rl) * DH;
    qa[0].u[0] = *(const uint4*)(qr + kq);
    qa[0].u[1] = *(const uint4*)(qr + kq + 16);
    qa[1].u[0] = *(const uint4*)(qr + 32 + kq);
    qa[1].u[1] = *(const uint4*)(qr + 32 + kq + 16);
  }

  v8f o[4];
  float mrow[8], lrow[8];
#pragma unroll
  for (int t = 0; t < 4; t++) o[t] = (v8f){0.f, 0.f, 0.f, 0.f, 0.f, 0.f, 0.f, 0.f};
#pragma unroll
  for (int i = 0; i < 8; i++) { mrow[i] = -3.0e38f; lrow[i] = 0.f; }

  const int kend = causal ? (q0 + 16) : N_;

  for (int n0 = 0; n0 < kend; n0 += 32) {
    // K^T B-frags: two 16-key halves x two 32-wide d slices.
    Frag kb00, kb01, kb10, kb11;
    {
      const __bf16* kr0 = Kp + (size_t)(n0 + rl) * DH;
      const __bf16* kr1 = Kp + (size_t)(n0 + 16 + rl) * DH;
      kb00.u[0] = *(const uint4*)(kr0 + kq);      kb00.u[1] = *(const uint4*)(kr0 + kq + 16);
      kb01.u[0] = *(const uint4*)(kr0 + 32 + kq); kb01.u[1] = *(const uint4*)(kr0 + 32 + kq + 16);
      kb10.u[0] = *(const uint4*)(kr1 + kq);      kb10.u[1] = *(const uint4*)(kr1 + kq + 16);
      kb11.u[0] = *(const uint4*)(kr1 + 32 + kq); kb11.u[1] = *(const uint4*)(kr1 + 32 + kq + 16);
    }

    v8f s0 = (v8f){0.f, 0.f, 0.f, 0.f, 0.f, 0.f, 0.f, 0.f};
    v8f s1 = (v8f){0.f, 0.f, 0.f, 0.f, 0.f, 0.f, 0.f, 0.f};
    s0 = WMMA_BF16(qa[0].f, kb00.f, s0);
    s0 = WMMA_BF16(qa[1].f, kb01.f, s0);
    s1 = WMMA_BF16(qa[0].f, kb10.f, s1);
    s1 = WMMA_BF16(qa[1].f, kb11.f, s1);

    const int key0 = n0 + rl, key1 = key0 + 16;
#pragma unroll
    for (int i = 0; i < 8; i++) {
      const int m = q0 + mro + i;
      float a0 = s0[i], a1 = s1[i];
      if (causal) {
        if (key0 > m) a0 = -3.0e38f;
        if (key1 > m) a1 = -3.0e38f;
      }
      // Row-max across the 16 lanes of this half (xor<16 stays in half).
      float r = fmaxf(a0, a1);
      r = fmaxf(r, __shfl_xor(r, 1));
      r = fmaxf(r, __shfl_xor(r, 2));
      r = fmaxf(r, __shfl_xor(r, 4));
      r = fmaxf(r, __shfl_xor(r, 8));
      const float nm  = fmaxf(mrow[i], r);
      const float fac = __expf(mrow[i] - nm);
      mrow[i] = nm;
      const float p0 = __expf(a0 - nm), p1 = __expf(a1 - nm);
      float sum = p0 + p1;
      sum += __shfl_xor(sum, 1);
      sum += __shfl_xor(sum, 2);
      sum += __shfl_xor(sum, 4);
      sum += __shfl_xor(sum, 8);
      lrow[i] = lrow[i] * fac + sum;
#pragma unroll
      for (int t = 0; t < 4; t++) o[t][i] *= fac;
      // Stash P row-major in LDS so it can be re-read in A-frag layout.
      const int prow = mro + i;
      Plds[prow * 32 + rl]      = (__bf16)p0;
      Plds[prow * 32 + rl + 16] = (__bf16)p1;
    }
    __syncthreads();  // single-wave block: HW NOP barrier, orders the LDS traffic

    Frag pa;
    pa.u[0] = *(const uint4*)&Plds[(size_t)rl * 32 + kq];
    pa.u[1] = *(const uint4*)&Plds[(size_t)rl * 32 + kq + 16];

#pragma unroll
    for (int t = 0; t < 4; t++) {
      Frag vb;
      const __bf16* vr = Vp + (size_t)(t * 16 + rl) * N_ + n0;
      vb.u[0] = *(const uint4*)(vr + kq);
      vb.u[1] = *(const uint4*)(vr + kq + 16);
      o[t] = WMMA_BF16(pa.f, vb.f, o[t]);
    }
    __syncthreads();
  }

  // Normalize and store as [b, n, h*64 + d].
#pragma unroll
  for (int i = 0; i < 8; i++) {
    const int gq = q0 + mro + i;
    const float inv = 1.f / lrow[i];
    const size_t rowbase = ((size_t)(b * N_ + gq)) * INNER + h * DH;
#pragma unroll
    for (int t = 0; t < 4; t++)
      O[rowbase + t * 16 + rl] = (__bf16)(o[t][i] * inv);
  }
}

// ---------------------------------------------------------------------------
// Launch
// ---------------------------------------------------------------------------
extern "C" void kernel_launch(void* const* d_in, const int* in_sizes, int n_in,
                              void* d_out, int out_size, void* d_ws, size_t ws_size,
                              hipStream_t stream) {
  const float* hs   = (const float*)d_in[0];  // [2,2048,1024]
  const float* Wqkv = (const float*)d_in[1];  // [1024,3072]
  const float* Wout = (const float*)d_in[2];  // [1024,1024]
  const float* bout = (const float*)d_in[3];  // [1024]
  const int*   isc  = (const int*)d_in[4];    // scalar
  float* out = (float*)d_out;

  // Workspace carve-up (bf16 elements), total 72 MiB.
  char* ws = (char*)d_ws;
  size_t off = 0;
  __bf16* Xb    = (__bf16*)(ws + off); off += (size_t)ROWS * QD * 2;      // 8 MiB
  __bf16* Wqkvt = (__bf16*)(ws + off); off += (size_t)QD * NQKV * 2;      // 6 MiB
  __bf16* Woutt = (__bf16*)(ws + off); off += (size_t)QD * INNER * 2;     // 2 MiB
  __bf16* qkvb  = (__bf16*)(ws + off); off += (size_t)ROWS * NQKV * 2;    // 24 MiB
  __bf16* Qb    = (__bf16*)(ws + off); off += (size_t)B_ * H_ * N_ * DH * 2;
  __bf16* Kb    = (__bf16*)(ws + off); off += (size_t)B_ * H_ * N_ * DH * 2;
  __bf16* Vtb   = (__bf16*)(ws + off); off += (size_t)B_ * H_ * N_ * DH * 2;
  __bf16* attnb = (__bf16*)(ws + off); off += (size_t)ROWS * INNER * 2;

  // 1) fp32 -> bf16 (X) and transposed bf16 weights.
  cvt_kernel<<<(ROWS * QD + 255) / 256, 256, 0, stream>>>(hs, Xb, ROWS * QD);
  cvt_t_kernel<<<(QD * NQKV + 255) / 256, 256, 0, stream>>>(Wqkv, Wqkvt, QD, NQKV);
  cvt_t_kernel<<<(QD * INNER + 255) / 256, 256, 0, stream>>>(Wout, Woutt, QD, INNER);

  // 2) QKV projection: [4096,1024] x [1024,3072] -> bf16 qkv.
  {
    dim3 g(NQKV / 128, ROWS / 128);
    gemm_bf16_kernel<0><<<g, 256, 0, stream>>>(Xb, Wqkvt, ROWS, NQKV, QD,
                                               nullptr, nullptr, qkvb);
  }

  // 3) Rotary + head split (+ V transpose), scale folded into Q.
  rotary_split_kernel<<<(B_ * H_ * N_ * (DH / 2) + 255) / 256, 256, 0, stream>>>(
      qkvb, Qb, Kb, Vtb);

  // 4) Causal flash attention: one wave per 16-row q tile.
  {
    dim3 g(N_ / 16, H_, B_);
    flash_attn_kernel<<<g, 32, 0, stream>>>(Qb, Kb, Vtb, attnb, isc);
  }

  // 5) Output projection + bias: [4096,1024] x [1024,1024] -> f32 out.
  {
    dim3 g(INNER / 128, ROWS / 128);
    gemm_bf16_kernel<1><<<g, 256, 0, stream>>>(attnb, Woutt, ROWS, INNER, QD,
                                               bout, out, nullptr);
  }
}
